// GNNStack_412316860635
// MI455X (gfx1250) — matrix operand order repaired and checked by reading, not compiled, not run on hardware
//
#include <hip/hip_runtime.h>

// ---------------------------------------------------------------------------
// GNNStack for MI455X (gfx1250, wave32, WMMA).
//
// Dead-code analysis of the reference: edge_attr / edge-update MLPs never
// influence the returned y. Eliminated. Remaining work:
//   deg -> 3x { scatter-mean ; relu(aggr/deg @ Wl + x @ Wr + b) } ->
//   post MLP -> gather-pair predict MLP -> y.
// GEMMs run on v_wmma_f32_16x16x32_f16 (f32 accumulate). Blocks own 64-row
// strips: 4 independent accumulators per wave, 4x reuse of each packed-B
// fragment, and all 4 A-fragments of a K-step are loaded before the WMMA
// burst so ds_load latency overlaps the matrix pipe. The scatter pass
// (~1.8 GB at 23.3 TB/s) is the roofline limiter.
// ---------------------------------------------------------------------------

typedef __attribute__((ext_vector_type(16))) _Float16 v16h;
typedef __attribute__((ext_vector_type(8)))  _Float16 v8h;
typedef __attribute__((ext_vector_type(4)))  _Float16 v4h;
typedef __attribute__((ext_vector_type(8)))  float    v8f;

#define DD 128

__device__ __forceinline__ v16h cat8(v8h lo, v8h hi) {
  return __builtin_shufflevector(lo, hi, 0,1,2,3,4,5,6,7,8,9,10,11,12,13,14,15);
}

__device__ __forceinline__ v4h cvt4(float4 f, float s) {
  v4h r;
  r[0] = (_Float16)(f.x * s);
  r[1] = (_Float16)(f.y * s);
  r[2] = (_Float16)(f.z * s);
  r[3] = (_Float16)(f.w * s);
  return r;
}

// ---------------- utility kernels ----------------
__global__ void k_zero(float* __restrict__ p, long n) {
  long i = (long)blockIdx.x * blockDim.x + threadIdx.x;
  if (i < n) p[i] = 0.0f;
}

__global__ void k_count_deg(const int* __restrict__ dst, float* __restrict__ deg, int E) {
  int e = blockIdx.x * blockDim.x + threadIdx.x;
  if (e < E) atomicAdd(&deg[dst[e]], 1.0f);
}

__global__ void k_finalize_rdeg(float* __restrict__ deg, int N) {
  int i = blockIdx.x * blockDim.x + threadIdx.x;
  if (i < N) deg[i] = 1.0f / fmaxf(deg[i], 1.0f);
}

// aggr[dst[e], :] += x[src[e], :]   (32 lanes per edge, float4 per lane)
__global__ void k_scatter(const float* __restrict__ x, const int* __restrict__ src,
                          const int* __restrict__ dst, float* __restrict__ aggr, int E) {
  long t = (long)blockIdx.x * blockDim.x + threadIdx.x;
  int e = (int)(t >> 5);
  if (e >= E) return;
  int c = (int)(t & 31) * 4;
  int s = src[e], d = dst[e];
  const float4 v = *(const float4*)(x + (long)s * DD + c);
  float* a = aggr + (long)d * DD + c;
  atomicAdd(a + 0, v.x);
  atomicAdd(a + 1, v.y);
  atomicAdd(a + 2, v.z);
  atomicAdd(a + 3, v.w);
}

// Pack a row-major f32 weight [K,128] into WMMA-B f16 tiles:
// out[((kt*8+nt)*32 + lane)*16 + e] = W[(kt*32 + (e&7) + 8*(lane>=16) + 16*(e>=8)) * 128 + nt*16 + (lane&15)]
// so each lane's 16 halves of a 32x16 B tile are contiguous (one 32-B load).
__global__ void k_pack_b(const float* __restrict__ W, _Float16* __restrict__ out, int K) {
  int idx = blockIdx.x * blockDim.x + threadIdx.x;
  if (idx >= K * DD) return;
  int e    = idx & 15;
  int lane = (idx >> 4) & 31;
  int nt   = (idx >> 9) & 7;
  int kt   = idx >> 12;
  int k = kt * 32 + (e & 7) + ((lane >= 16) ? 8 : 0) + ((e >= 8) ? 16 : 0);
  int n = nt * 16 + (lane & 15);
  out[idx] = (_Float16)W[k * DD + n];
}

// ---------------- WMMA node GEMM ----------------
// out = act( A1*rscale @ W1 [+ A2 @ W2] + bias ), A [rows,128], W [128,128].
// One block = 64-row strip (4 M-tiles); 8 waves = 8 column tiles.
template <bool DUAL>
__global__ __launch_bounds__(256)
void k_node_gemm(const float* __restrict__ A1, const float* __restrict__ rscale,
                 const _Float16* __restrict__ W1p,
                 const float* __restrict__ A2, const _Float16* __restrict__ W2p,
                 const float* __restrict__ bias, float* __restrict__ out,
                 int rows, int do_relu) {
  __shared__ __align__(16) _Float16 sA[DUAL ? 2 : 1][64 * DD];
  const int row0 = blockIdx.x * 64;

  // stage 64x128 (x2) as f16, float4-vectorized
  for (int idx = threadIdx.x; idx < 64 * 32; idx += blockDim.x) {
    int m  = idx >> 5;
    int k4 = (idx & 31) << 2;
    int r = row0 + m; if (r >= rows) r = rows - 1;   // clamped rows never stored
    float s = rscale ? rscale[r] : 1.0f;
    float4 a1 = *(const float4*)(A1 + (long)r * DD + k4);
    *(v4h*)&sA[0][m * DD + k4] = cvt4(a1, s);
    if (DUAL) {
      float4 a2 = *(const float4*)(A2 + (long)r * DD + k4);
      *(v4h*)&sA[1][m * DD + k4] = cvt4(a2, 1.0f);
    }
  }
  __syncthreads();

  const unsigned lane = threadIdx.x & 31u;
  const int wave = threadIdx.x >> 5;        // column tile 0..7
  const int m    = lane & 15;               // A row within M-tile for this lane
  const int hi16 = (lane >= 16) ? 1 : 0;

  v8f acc[4];
#pragma unroll
  for (int mt = 0; mt < 4; ++mt) acc[mt] = (v8f){0.f,0.f,0.f,0.f,0.f,0.f,0.f,0.f};

#pragma unroll
  for (int kb = 0; kb < 4; ++kb) {
    const int k1 = kb * 32 + hi16 * 8;      // A layout: runs {k1..k1+7},{k1+16..k1+23}
    const int boff = ((kb * 8 + wave) * 32 + (int)lane) << 4;
    v16h b1 = *(const v16h*)(W1p + boff);
    v16h a[4];
    // bank 0: load all 4 fragments, then the WMMA burst (waits stagger/overlap)
#pragma unroll
    for (int mt = 0; mt < 4; ++mt) {
      const int arow = (mt * 16 + m) * DD + k1;
      a[mt] = cat8(*(const v8h*)&sA[0][arow], *(const v8h*)&sA[0][arow + 16]);
    }
#pragma unroll
    for (int mt = 0; mt < 4; ++mt)
      acc[mt] = __builtin_amdgcn_wmma_f32_16x16x32_f16(false, a[mt], false, b1,
                                                       (short)0, acc[mt], false, false);
    if (DUAL) {
      v16h b2 = *(const v16h*)(W2p + boff);
#pragma unroll
      for (int mt = 0; mt < 4; ++mt) {
        const int arow = (mt * 16 + m) * DD + k1;
        a[mt] = cat8(*(const v8h*)&sA[1][arow], *(const v8h*)&sA[1][arow + 16]);
      }
#pragma unroll
      for (int mt = 0; mt < 4; ++mt)
        acc[mt] = __builtin_amdgcn_wmma_f32_16x16x32_f16(false, a[mt], false, b2,
                                                         (short)0, acc[mt], false, false);
    }
  }

  const int col = wave * 16 + (lane & 15);
  const float b = bias ? bias[col] : 0.0f;
#pragma unroll
  for (int mt = 0; mt < 4; ++mt) {
#pragma unroll
    for (int r = 0; r < 8; ++r) {
      int row = row0 + mt * 16 + r + hi16 * 8;   // C/D: VGPR r holds M=r (+8 hi lanes)
      if (row < rows) {
        float v = acc[mt][r] + b;
        if (do_relu) v = fmaxf(v, 0.0f);
        out[(long)row * DD + col] = v;
      }
    }
  }
}

// ---------------- predict hidden: t = relu([x[i], x[j]] @ qW1 + qb1) ----------------
// 64 edges per block (4 M-tiles), K = 256 (concat of the two gathered rows).
__global__ __launch_bounds__(256)
void k_predict_hidden(const float* __restrict__ x2, const int* __restrict__ pi,
                      const int* __restrict__ pj, const _Float16* __restrict__ Wp,
                      const float* __restrict__ bias, float* __restrict__ t, int EP) {
  __shared__ __align__(16) _Float16 sA[64 * 256];
  const int e0 = blockIdx.x * 64;

  for (int idx = threadIdx.x; idx < 64 * 64; idx += blockDim.x) {
    int m  = idx >> 6;
    int k4 = (idx & 63) << 2;                    // 0..252, 4-aligned -> same half
    int e = e0 + m; if (e >= EP) e = EP - 1;
    int row = (k4 < DD) ? pi[e] : pj[e];
    float4 a = *(const float4*)(x2 + (long)row * DD + (k4 & 127));
    *(v4h*)&sA[m * 256 + k4] = cvt4(a, 1.0f);
  }
  __syncthreads();

  const unsigned lane = threadIdx.x & 31u;
  const int wave = threadIdx.x >> 5;
  const int m    = lane & 15;
  const int hi16 = (lane >= 16) ? 1 : 0;

  v8f acc[4];
#pragma unroll
  for (int mt = 0; mt < 4; ++mt) acc[mt] = (v8f){0.f,0.f,0.f,0.f,0.f,0.f,0.f,0.f};

#pragma unroll
  for (int kb = 0; kb < 8; ++kb) {
    const int k1 = kb * 32 + hi16 * 8;
    v16h b = *(const v16h*)(Wp + (((kb * 8 + wave) * 32 + (int)lane) << 4));
    v16h a[4];
#pragma unroll
    for (int mt = 0; mt < 4; ++mt) {
      const int arow = (mt * 16 + m) * 256 + k1;
      a[mt] = cat8(*(const v8h*)&sA[arow], *(const v8h*)&sA[arow + 16]);
    }
#pragma unroll
    for (int mt = 0; mt < 4; ++mt)
      acc[mt] = __builtin_amdgcn_wmma_f32_16x16x32_f16(false, a[mt], false, b,
                                                       (short)0, acc[mt], false, false);
  }

  const int col = wave * 16 + (lane & 15);
  const float bb = bias[col];
#pragma unroll
  for (int mt = 0; mt < 4; ++mt) {
#pragma unroll
    for (int r = 0; r < 8; ++r) {
      int row = e0 + mt * 16 + r + hi16 * 8;
      if (row < EP) t[(long)row * DD + col] = fmaxf(acc[mt][r] + bb, 0.0f);
    }
  }
}

// ---------------- final score: y = t @ qW2 + qb2 ----------------
__global__ void k_score(const float* __restrict__ t, const float* __restrict__ w,
                        const float* __restrict__ b, float* __restrict__ y, int EP) {
  int e = blockIdx.x * blockDim.x + threadIdx.x;
  if (e >= EP) return;
  const float4* tv = (const float4*)(t + (long)e * DD);
  float acc = 0.0f;
#pragma unroll
  for (int i = 0; i < 32; ++i) {
    float4 a  = tv[i];
    float4 ww = *(const float4*)(w + i * 4);
    acc += a.x * ww.x + a.y * ww.y + a.z * ww.z + a.w * ww.w;
  }
  y[e] = acc + b[0];
}

// ---------------------------------------------------------------------------
extern "C" void kernel_launch(void* const* d_in, const int* in_sizes, int n_in,
                              void* d_out, int out_size, void* d_ws, size_t ws_size,
                              hipStream_t stream) {
  const float* x    = (const float*)d_in[0];
  const int*   eidx = (const int*)d_in[2];
  const int*   pidx = (const int*)d_in[3];
  const float* Wl   = (const float*)d_in[4];
  const float* bl   = (const float*)d_in[5];
  const float* Wr   = (const float*)d_in[6];
  // d_in[7..10]: edge-update MLP params -- dead code w.r.t. output, unused.
  const float* pW1  = (const float*)d_in[11];
  const float* pb1  = (const float*)d_in[12];
  const float* pW2  = (const float*)d_in[13];
  const float* pb2  = (const float*)d_in[14];
  const float* qW1  = (const float*)d_in[15];
  const float* qb1  = (const float*)d_in[16];
  const float* qW2  = (const float*)d_in[17];
  const float* qb2  = (const float*)d_in[18];

  const int N  = in_sizes[0] / DD;
  const int E  = in_sizes[1];
  const int EP = in_sizes[3] / 2;
  const int L  = in_sizes[4] / (DD * DD);

  // ---- workspace carve (256-B aligned) ----
  size_t off = 0;
  auto carve = [&](size_t bytes) -> char* {
    char* p = (char*)d_ws + off;
    off = (off + bytes + 255) & ~(size_t)255;
    return p;
  };
  float*    rdeg = (float*)carve((size_t)N * 4);
  float*    aggr = (float*)carve((size_t)N * DD * 4);
  float*    xA   = (float*)carve((size_t)N * DD * 4);
  float*    xB   = (float*)carve((size_t)N * DD * 4);
  float*    tbuf = (float*)carve((size_t)EP * DD * 4);
  _Float16* WlP  = (_Float16*)carve((size_t)L * DD * DD * 2);
  _Float16* WrP  = (_Float16*)carve((size_t)L * DD * DD * 2);
  _Float16* pW1P = (_Float16*)carve((size_t)DD * DD * 2);
  _Float16* pW2P = (_Float16*)carve((size_t)DD * DD * 2);
  _Float16* qW1P = (_Float16*)carve((size_t)2 * DD * DD * 2);

  const int* src = eidx;        // edge_index[0]
  const int* dst = eidx + E;    // edge_index[1]
  const int* pi  = pidx;        // predict_edge_index[0]
  const int* pj  = pidx + EP;   // predict_edge_index[1]

  const int TB = 256;
  const int gW = (DD * DD + TB - 1) / TB;

  // pack all weights into WMMA-B f16 tiles (tiny, once per call)
  for (int l = 0; l < L; ++l) {
    k_pack_b<<<gW, TB, 0, stream>>>(Wl + (size_t)l * DD * DD, WlP + (size_t)l * DD * DD, DD);
    k_pack_b<<<gW, TB, 0, stream>>>(Wr + (size_t)l * DD * DD, WrP + (size_t)l * DD * DD, DD);
  }
  k_pack_b<<<gW, TB, 0, stream>>>(pW1, pW1P, DD);
  k_pack_b<<<gW, TB, 0, stream>>>(pW2, pW2P, DD);
  k_pack_b<<<(2 * DD * DD + TB - 1) / TB, TB, 0, stream>>>(qW1, qW1P, 2 * DD);

  // degrees -> 1/max(deg,1)
  k_zero<<<(N + TB - 1) / TB, TB, 0, stream>>>(rdeg, N);
  k_count_deg<<<(E + TB - 1) / TB, TB, 0, stream>>>(dst, rdeg, E);
  k_finalize_rdeg<<<(N + TB - 1) / TB, TB, 0, stream>>>(rdeg, N);

  // SAGE layers
  const long ND = (long)N * DD;
  const int gemmGrid = (N + 63) / 64;
  const float* xcur = x;
  float* xnext = xA;
  for (int l = 0; l < L; ++l) {
    k_zero<<<(int)((ND + TB - 1) / TB), TB, 0, stream>>>(aggr, ND);
    k_scatter<<<(int)(((long)E * 32 + TB - 1) / TB), TB, 0, stream>>>(xcur, src, dst, aggr, E);
    k_node_gemm<true><<<gemmGrid, TB, 0, stream>>>(
        aggr, rdeg, WlP + (size_t)l * DD * DD,
        xcur, WrP + (size_t)l * DD * DD,
        bl + (size_t)l * DD, xnext, N, /*relu=*/1);
    xcur  = xnext;
    xnext = (xnext == xA) ? xB : xA;
  }

  // node post MLP
  float* h1 = xnext;
  float* h2 = (h1 == xA) ? xB : xA;
  k_node_gemm<false><<<gemmGrid, TB, 0, stream>>>(xcur, nullptr, pW1P, nullptr, nullptr,
                                                  pb1, h1, N, /*relu=*/1);
  k_node_gemm<false><<<gemmGrid, TB, 0, stream>>>(h1, nullptr, pW2P, nullptr, nullptr,
                                                  pb2, h2, N, /*relu=*/0);

  // edge prediction
  k_predict_hidden<<<(EP + 63) / 64, TB, 0, stream>>>(h2, pi, pj, qW1P, qb1, tbuf, EP);
  k_score<<<(EP + TB - 1) / TB, TB, 0, stream>>>(tbuf, qW2, qb2, (float*)d_out, EP);
}
